// ClassificationLoss_33423435497885
// MI455X (gfx1250) — compile-verified, matrix operand order checked by reading, not run on hardware
//
#include <hip/hip_runtime.h>
#include <hip/hip_bf16.h>

typedef __attribute__((ext_vector_type(16))) _Float16 v16h;
typedef __attribute__((ext_vector_type(8)))  float    v8f;

constexpr int B_IMG   = 16;
constexpr int N_ELEM  = 1 << 20;          // 1024*1024 per image
constexpr int HARD    = 512;
constexpr int RANDN   = 512;
constexpr int NB      = 65536;            // 16-bit value bins
constexpr int U_LEN   = N_ELEM - HARD;    // rand_u length per image
constexpr int CHUNK   = 4096;             // bins per scan block
constexpr int NCHUNK  = NB / CHUNK;       // 16
constexpr int CANDMAX = 4096;

struct Stats {
  unsigned num_pos;
  float    pos_loss;
  float    hard_loss;
  float    rand_loss;
  unsigned u_t;       // u threshold bin
  unsigned u_below;   // count of valid u strictly below bin u_t
  unsigned u_target;  // min(512, num_neg-512)
  unsigned cand_cnt;
  unsigned sel_fill;
  unsigned sel_cnt;
  unsigned pad[6];    // 64B
};

struct Cand { float u; int j; };

__device__ __forceinline__ unsigned conf_key(float v) {
  // descending key: 0 = largest value bin
  float q = v * 65536.0f;
  unsigned k = (q >= 65535.0f) ? 65535u : (unsigned)fmaxf(q, 0.0f);
  return 65535u - k;
}
__device__ __forceinline__ unsigned u_key(float v) {
  float q = v * 65536.0f;
  return (q >= 65535.0f) ? 65535u : (unsigned)fmaxf(q, 0.0f);
}
__device__ __forceinline__ float log_loss_1m(float v) {
  // -clip(log(1-v), -100) ; log(0) -> clamp at 100
  return fminf(-logf(1.0f - v), 100.0f);
}

// ---------------------------------------------------------------------------
// 256-value inclusive scan + 16 column offsets, one full wave32, 4 WMMAs.
//   S   = L    * X : L lower-triangular ones -> per-column inclusive scans
//   OFF = Lst  * T : strictly-lower A over column totals -> exclusive offsets
// Exactness: counts are split v = 2048*hi + lo; hi,lo < 2048 are exact in
// f16, accumulation is f32 (integers < 2^24 exact), recombined as lo+2048*hi.
// EXEC is all ones for the executing wave (guard is wave-uniform).
// Same-wave LDS RAW (sc write -> sc read) is safe: DS ops are in-order per
// wave and the compiler inserts s_wait_dscnt for the load return.
// ---------------------------------------------------------------------------
__device__ __forceinline__ void wmma_scan256(const float* __restrict__ tot,
                                             float* __restrict__ sc,
                                             float* __restrict__ offs) {
  const int l = threadIdx.x & 31;
  const int m = l & 15;                 // A row / B column
  // element e -> K mapping for 16-bit A (16x32) / B (32x16) operands
  int Ke[16];
#pragma unroll
  for (int e = 0; e < 16; ++e) {
    const int vg = e >> 1, h = e & 1;
    Ke[e] = (vg < 4) ? (((l < 16) ? 0 : 8) + vg * 2 + h)
                     : (((l < 16) ? 16 : 24) + (vg - 4) * 2 + h);
  }
  v16h aInc, aExc, bLo, bHi;
#pragma unroll
  for (int e = 0; e < 16; ++e) {
    const int K = Ke[e];
    aInc[e] = (_Float16)((K <= m) ? 1.0f : 0.0f);   // K>=16 -> 0
    aExc[e] = (_Float16)((K <  m) ? 1.0f : 0.0f);
    if (K < 16) {
      const float v = tot[m * 16 + K];
      const float hi = floorf(v * (1.0f / 2048.0f));
      bLo[e] = (_Float16)(v - hi * 2048.0f);
      bHi[e] = (_Float16)hi;
    } else {
      bLo[e] = (_Float16)0.0f;
      bHi[e] = (_Float16)0.0f;
    }
  }
  v8f cLo = {}, cHi = {};
  cLo = __builtin_amdgcn_wmma_f32_16x16x32_f16(false, aInc, false, bLo,
                                               (short)0, cLo, false, false);
  cHi = __builtin_amdgcn_wmma_f32_16x16x32_f16(false, aInc, false, bHi,
                                               (short)0, cHi, false, false);
#pragma unroll
  for (int v = 0; v < 8; ++v) {
    const int row = v + ((l < 16) ? 0 : 8);
    sc[m * 16 + row] = cLo[v] + 2048.0f * cHi[v];   // inclusive scan, col m
  }
  // exclusive scan of 16 column totals T[k] = sc[k*16+15]
  v16h tLo, tHi;
#pragma unroll
  for (int e = 0; e < 16; ++e) {
    const int K = Ke[e];
    if (K < 16) {
      const float v = sc[K * 16 + 15];
      const float hi = floorf(v * (1.0f / 2048.0f));
      tLo[e] = (_Float16)(v - hi * 2048.0f);
      tHi[e] = (_Float16)hi;
    } else {
      tLo[e] = (_Float16)0.0f;
      tHi[e] = (_Float16)0.0f;
    }
  }
  v8f oLo = {}, oHi = {};
  oLo = __builtin_amdgcn_wmma_f32_16x16x32_f16(false, aExc, false, tLo,
                                               (short)0, oLo, false, false);
  oHi = __builtin_amdgcn_wmma_f32_16x16x32_f16(false, aExc, false, tHi,
                                               (short)0, oHi, false, false);
#pragma unroll
  for (int v = 0; v < 8; ++v) {
    const int row = v + ((l < 16) ? 0 : 8);
    if (m == 0) offs[row] = oLo[v] + 2048.0f * oHi[v];  // column 0 suffices
  }
}

__global__ void k_zero(unsigned* __restrict__ p, size_t nwords) {
  size_t i = (size_t)blockIdx.x * blockDim.x + threadIdx.x;
  const size_t stride = (size_t)gridDim.x * blockDim.x;
  for (; i < nwords; i += stride) p[i] = 0u;
}

// pass 1: positive stats + negative-conf histogram
__global__ void k_hist_conf(const unsigned char* __restrict__ pos,
                            const float* __restrict__ conf,
                            unsigned* __restrict__ hist,
                            Stats* __restrict__ stats) {
  const int img = blockIdx.y;
  __shared__ unsigned s_np;
  __shared__ float s_pl;
  if (threadIdx.x == 0) { s_np = 0u; s_pl = 0.0f; }
  __syncthreads();
  const size_t base = (size_t)img * N_ELEM;
  const size_t stride = (size_t)gridDim.x * blockDim.x;
  unsigned np = 0; float pl = 0.0f;
  for (size_t i = (size_t)blockIdx.x * blockDim.x + threadIdx.x; i < (size_t)N_ELEM; i += stride) {
    const float v = conf[base + i];
    if (pos[base + i]) {
      np += 1u;
      pl += fminf(-logf(v), 100.0f);
    } else {
      atomicAdd(&hist[(size_t)img * NB + conf_key(v)], 1u);
    }
  }
  atomicAdd(&s_np, np);
  atomicAdd(&s_pl, pl);
  __syncthreads();
  if (threadIdx.x == 0) {
    atomicAdd(&stats[img].num_pos, s_np);
    atomicAdd(&stats[img].pos_loss, s_pl);
  }
}

// block scan of a 4096-bin chunk; wave 0 does the 256-wide scan via WMMA
__global__ void k_scan_chunks(const unsigned* __restrict__ hist,
                              unsigned* __restrict__ cdf,
                              unsigned* __restrict__ chunkSum) {
  __shared__ float tot[256];
  __shared__ float sc[256];
  __shared__ float offs[16];
  const int img = blockIdx.y, chunk = blockIdx.x, t = threadIdx.x;
  const size_t off = (size_t)img * NB + (size_t)chunk * CHUNK + (size_t)t * 16;
  const unsigned* h = hist + off;
  unsigned loc[16];
  unsigned run = 0;
#pragma unroll
  for (int i = 0; i < 16; ++i) { run += h[i]; loc[i] = run; }
  tot[t] = (float)run;
  __syncthreads();
  if (t < 32) wmma_scan256(tot, sc, offs);
  __syncthreads();
  const float excl = offs[t >> 4] + sc[t] - tot[t];
  const unsigned basec = (unsigned)(excl + 0.5f);
  unsigned* o = cdf + off;
#pragma unroll
  for (int i = 0; i < 16; ++i) o[i] = basec + loc[i];
  if (t == 0) {
    chunkSum[img * NCHUNK + chunk] = (unsigned)(offs[15] + sc[255] + 0.5f);
  }
}

__global__ void k_scan_tops(const unsigned* __restrict__ chunkSum,
                            unsigned* __restrict__ chunkOff) {
  const int img = threadIdx.x;
  if (img >= B_IMG) return;
  unsigned run = 0;
  for (int c = 0; c < NCHUNK; ++c) {
    chunkOff[img * NCHUNK + c] = run;
    run += chunkSum[img * NCHUNK + c];
  }
}

__global__ void k_scan_add(unsigned* __restrict__ cdf,
                           const unsigned* __restrict__ chunkOff,
                           const unsigned* __restrict__ hist,
                           unsigned* __restrict__ cursor) {
  const size_t i = (size_t)blockIdx.x * blockDim.x + threadIdx.x;
  if (i >= (size_t)B_IMG * NB) return;
  const int img = (int)(i >> 16);
  const int b = (int)(i & 65535u);
  const unsigned v = cdf[i] + chunkOff[img * NCHUNK + (b / CHUNK)];
  cdf[i] = v;
  if (cursor) cursor[i] = v - hist[i];   // exclusive prefix = bin start
}

// counting-sort scatter of negatives into bin-contiguous bucket array
__global__ void k_scatter(const unsigned char* __restrict__ pos,
                          const float* __restrict__ conf,
                          unsigned* __restrict__ cursor,
                          float* __restrict__ bucket) {
  const int img = blockIdx.y;
  const size_t base = (size_t)img * N_ELEM;
  const size_t stride = (size_t)gridDim.x * blockDim.x;
  for (size_t i = (size_t)blockIdx.x * blockDim.x + threadIdx.x; i < (size_t)N_ELEM; i += stride) {
    if (pos[base + i]) continue;
    const float v = conf[base + i];
    const unsigned p = atomicAdd(&cursor[(size_t)img * NB + conf_key(v)], 1u);
    bucket[base + p] = v;
  }
}

__global__ void k_hist_u(const float* __restrict__ u,
                         unsigned* __restrict__ uhist,
                         const Stats* __restrict__ stats) {
  const int img = blockIdx.y;
  long valid = (long)N_ELEM - (long)stats[img].num_pos - HARD;
  if (valid < 0) valid = 0;
  const size_t base = (size_t)img * U_LEN;
  const size_t stride = (size_t)gridDim.x * blockDim.x;
  for (size_t j = (size_t)blockIdx.x * blockDim.x + threadIdx.x; j < (size_t)U_LEN; j += stride) {
    if ((long)j >= valid) continue;
    atomicAdd(&uhist[(size_t)img * NB + u_key(u[base + j])], 1u);
  }
}

__global__ void k_usetup(const unsigned* __restrict__ ucdf, Stats* __restrict__ stats) {
  const int img = threadIdx.x;
  if (img >= B_IMG) return;
  long valid = (long)N_ELEM - (long)stats[img].num_pos - HARD;
  if (valid < 0) valid = 0;
  const unsigned target = (unsigned)((valid < RANDN) ? valid : RANDN);
  stats[img].u_target = target;
  if (target == 0) { stats[img].u_t = 0; stats[img].u_below = 0; return; }
  const unsigned* c = ucdf + (size_t)img * NB;
  int lo = 0, hi = NB - 1;                 // smallest bin with cum >= target
  while (lo < hi) { const int mid = (lo + hi) >> 1; if (c[mid] >= target) hi = mid; else lo = mid + 1; }
  stats[img].u_t = (unsigned)lo;
  stats[img].u_below = lo ? c[lo - 1] : 0u;
}

__global__ void k_ucollect(const float* __restrict__ u, Stats* __restrict__ stats,
                           int* __restrict__ sel, Cand* __restrict__ cand) {
  const int img = blockIdx.y;
  const unsigned target = stats[img].u_target;
  if (target == 0) return;
  const unsigned t = stats[img].u_t;
  long valid = (long)N_ELEM - (long)stats[img].num_pos - HARD;
  if (valid < 0) valid = 0;
  const size_t base = (size_t)img * U_LEN;
  const size_t stride = (size_t)gridDim.x * blockDim.x;
  for (size_t j = (size_t)blockIdx.x * blockDim.x + threadIdx.x; j < (size_t)U_LEN; j += stride) {
    if ((long)j >= valid) continue;
    const float v = u[base + j];
    const unsigned k = u_key(v);
    if (k < t) {
      const unsigned s = atomicAdd(&stats[img].sel_fill, 1u);
      if (s < (unsigned)RANDN) sel[img * RANDN + s] = (int)j;
    } else if (k == t) {
      const unsigned c = atomicAdd(&stats[img].cand_cnt, 1u);
      if (c < (unsigned)CANDMAX) {
        cand[(size_t)img * CANDMAX + c].u = v;
        cand[(size_t)img * CANDMAX + c].j = (int)j;
      }
    }
  }
}

// stable-order rank among threshold-bin candidates (matches stable argsort ties)
__global__ void k_ucand_pick(const Cand* __restrict__ cand, Stats* __restrict__ stats,
                             int* __restrict__ sel) {
  const int img = blockIdx.x;
  const unsigned target = stats[img].u_target;
  if (target == 0) { if (threadIdx.x == 0) stats[img].sel_cnt = 0u; return; }
  const unsigned below = stats[img].u_below;
  const unsigned need = target - below;
  unsigned m = stats[img].cand_cnt;
  if (m > (unsigned)CANDMAX) m = CANDMAX;
  const Cand* c = cand + (size_t)img * CANDMAX;
  for (unsigned a = threadIdx.x; a < m; a += blockDim.x) {
    const float va = c[a].u; const int ja = c[a].j;
    unsigned rank = 0;
    for (unsigned b = 0; b < m; ++b) {
      const float vb = c[b].u; const int jb = c[b].j;
      rank += (unsigned)((vb < va) || (vb == va && jb < ja));
    }
    if (rank < need) sel[img * RANDN + below + rank] = ja;
  }
  if (threadIdx.x == 0) {
    const unsigned got = (need < m) ? need : m;
    stats[img].sel_cnt = below + got;
  }
}

// rank-indexed value lookup: binary search CDF, tie-aware select inside bin
__device__ __forceinline__ float rank_value(const unsigned* __restrict__ cdf,
                                            const float* __restrict__ bucket,
                                            int img, unsigned r) {
  const unsigned* c = cdf + (size_t)img * NB;
  int lo = 0, hi = NB - 1;
  const unsigned want = r + 1;
  while (lo < hi) { const int mid = (lo + hi) >> 1; if (c[mid] >= want) hi = mid; else lo = mid + 1; }
  const unsigned start = lo ? c[lo - 1] : 0u;
  const unsigned cnt = c[lo] - start;
  const unsigned kk = r - start;                 // local descending rank
  const float* bin = bucket + (size_t)img * N_ELEM + start;
  float val = 0.5f;
  for (unsigned a = 0; a < cnt; ++a) {
    const float va = bin[a];
    unsigned g = 0, e = 0;
    for (unsigned b = 0; b < cnt; ++b) {
      const float vb = bin[b];
      g += (unsigned)(vb > va);
      e += (unsigned)(vb == va);
    }
    if (g <= kk && kk < g + e) { val = va; break; }
  }
  return val;
}

__global__ void k_hard_rand(const unsigned* __restrict__ cdf,
                            const float* __restrict__ bucket,
                            const int* __restrict__ sel,
                            Stats* __restrict__ stats) {
  const int img = blockIdx.x;
  const int t = threadIdx.x;                     // 0..1023
  const unsigned num_neg = (unsigned)(N_ELEM - (long)stats[img].num_pos);
  __shared__ float sh, sr;
  if (t == 0) { sh = 0.0f; sr = 0.0f; }
  __syncthreads();
  const bool isHard = t < HARD;
  long r = -1;
  if (isHard) {
    if ((unsigned)t < num_neg) r = t;            // hard_valid
  } else {
    const int s = t - HARD;
    if ((unsigned)s < stats[img].sel_cnt) {
      const long rr = (long)HARD + (long)sel[img * RANDN + s];
      if (rr < (long)num_neg) r = rr;
    }
  }
  if (r >= 0) {
    const float v = rank_value(cdf, bucket, img, (unsigned)r);
    const float contrib = log_loss_1m(v);
    if (isHard) atomicAdd(&sh, contrib); else atomicAdd(&sr, contrib);
  }
  __syncthreads();
  if (t == 0) {
    atomicAdd(&stats[img].hard_loss, sh);
    atomicAdd(&stats[img].rand_loss, sr);
  }
}

__global__ void k_finalize(const Stats* __restrict__ stats, float* __restrict__ out) {
  const int img = threadIdx.x;
  if (img >= B_IMG) return;
  const Stats s = stats[img];
  const long num_neg = (long)N_ELEM - (long)s.num_pos;
  const bool has_rem = (num_neg - HARD) > 0;
  const float sample = (float)s.num_pos + (float)((num_neg < HARD) ? num_neg : (long)HARD);
  const float full = s.pos_loss + s.hard_loss + s.rand_loss;
  const float alt = (s.pos_loss + s.hard_loss) / fmaxf(sample, 1.0f);
  out[img] = has_rem ? full : alt;
}

extern "C" void kernel_launch(void* const* d_in, const int* in_sizes, int n_in,
                              void* d_out, int out_size, void* d_ws, size_t ws_size,
                              hipStream_t stream) {
  (void)in_sizes; (void)n_in; (void)out_size; (void)ws_size;
  const unsigned char* pos = (const unsigned char*)d_in[0];  // bool (1B)
  const float* conf = (const float*)d_in[1];
  const float* u = (const float*)d_in[2];
  float* out = (float*)d_out;

  // bump allocator over d_ws (~85 MB total)
  char* ws = (char*)d_ws;
  size_t off = 0;
  auto alloc = [&](size_t bytes) -> char* {
    char* p = ws + off;
    off = (off + bytes + 255) & ~(size_t)255;
    return p;
  };
  unsigned* hist   = (unsigned*)alloc((size_t)B_IMG * NB * 4);      // zeroed
  unsigned* uhist  = (unsigned*)alloc((size_t)B_IMG * NB * 4);      // zeroed
  Stats*    stats  = (Stats*)   alloc(sizeof(Stats) * B_IMG);       // zeroed
  const size_t zero_words = off / 4;
  unsigned* cdf    = (unsigned*)alloc((size_t)B_IMG * NB * 4);
  unsigned* ucdf   = (unsigned*)alloc((size_t)B_IMG * NB * 4);
  unsigned* cursor = (unsigned*)alloc((size_t)B_IMG * NB * 4);
  unsigned* chunkS = (unsigned*)alloc((size_t)B_IMG * NCHUNK * 4);
  unsigned* chunkO = (unsigned*)alloc((size_t)B_IMG * NCHUNK * 4);
  Cand*     cand   = (Cand*)    alloc(sizeof(Cand) * B_IMG * CANDMAX);
  int*      sel    = (int*)     alloc(sizeof(int) * B_IMG * RANDN);
  float*    bucket = (float*)   alloc((size_t)B_IMG * N_ELEM * 4);

  const dim3 gN(512, B_IMG);
  const unsigned scanAddBlocks = (unsigned)(((size_t)B_IMG * NB + 255) / 256);

  k_zero<<<1024, 256, 0, stream>>>((unsigned*)d_ws, zero_words);
  k_hist_conf<<<gN, 256, 0, stream>>>(pos, conf, hist, stats);
  k_scan_chunks<<<dim3(NCHUNK, B_IMG), 256, 0, stream>>>(hist, cdf, chunkS);
  k_scan_tops<<<1, B_IMG, 0, stream>>>(chunkS, chunkO);
  k_scan_add<<<scanAddBlocks, 256, 0, stream>>>(cdf, chunkO, hist, cursor);
  k_scatter<<<gN, 256, 0, stream>>>(pos, conf, cursor, bucket);
  k_hist_u<<<gN, 256, 0, stream>>>(u, uhist, stats);
  k_scan_chunks<<<dim3(NCHUNK, B_IMG), 256, 0, stream>>>(uhist, ucdf, chunkS);
  k_scan_tops<<<1, B_IMG, 0, stream>>>(chunkS, chunkO);
  k_scan_add<<<scanAddBlocks, 256, 0, stream>>>(ucdf, chunkO, uhist, nullptr);
  k_usetup<<<1, B_IMG, 0, stream>>>(ucdf, stats);
  k_ucollect<<<gN, 256, 0, stream>>>(u, stats, sel, cand);
  k_ucand_pick<<<B_IMG, 256, 0, stream>>>(cand, stats, sel);
  k_hard_rand<<<B_IMG, 1024, 0, stream>>>(cdf, bucket, sel, stats);
  k_finalize<<<1, B_IMG, 0, stream>>>(stats, out);
}
